// EngramLayer_23476291240536
// MI455X (gfx1250) — matrix operand order repaired and checked by reading, not compiled, not run on hardware
//
#include <hip/hip_runtime.h>
#include <hip/hip_bf16.h>

// ---------------------------------------------------------------------------
// EngramLayer for MI455X (gfx1250, wave32, WMMA bf16 + TDM async tile DMA)
//   k_cvt        : Wk/Wv/Wo f32 -> bf16 (one-time, weights live in L2)
//   k_embed_conv : hash-gather + dilated depthwise conv + LN + SiLU -> E(bf16)
//   k_kv_gate    : key GEMM (WMMA) -> LN gate -> value GEMM (WMMA) -> Vg(bf16)
//   k_out_gemm   : Vg @ Wo^T, 128x128 tiles, TENSOR_LOAD_TO_LDS double-buffer
// ---------------------------------------------------------------------------

typedef __attribute__((ext_vector_type(16))) __bf16 v16b;
typedef __attribute__((ext_vector_type(8)))  float  v8f;
typedef unsigned int u32x4 __attribute__((ext_vector_type(4)));
typedef int          i32x8 __attribute__((ext_vector_type(8)));
typedef int          i32x4 __attribute__((ext_vector_type(4)));

#define BATCH   4
#define SEQ     4096
#define BT      (BATCH * SEQ)      // 16384 rows
#define HID     2048
#define EDIM    256
#define NHEADS  8
#define HDIM    32
#define TCHUNK  32
#define HALO    9                  // (KERNEL-1)*DIL = 3*3

// ---- WMMA fragment loaders (16x16x32 bf16) --------------------------------
// A (16x32, MxK): lane row = lane&15; holds K = kb..kb+15,
//                 kb = 16*(lane>=16). Two 16B contiguous loads.
__device__ __forceinline__ v16b load_a16x32(const __bf16* tile, int ld, int koff, int lane) {
  int row = lane & 15;
  int kb  = koff + ((lane >> 4) << 4);
  const __bf16* p = tile + (size_t)row * ld + kb;
  union { v16b v; uint4 u[2]; } r;
  r.u[0] = *(const uint4*)(p);
  r.u[1] = *(const uint4*)(p + 8);
  return r.v;
}

// B (32x16, KxN) with B[k,n] = W[n,k] (W row-major, K contiguous):
// lane col = lane&15; holds 16 contiguous K starting at 16*(lane>=16).
// Caller passes p = W + n_row*ldw + k0 + 16*(lane>=16).
__device__ __forceinline__ v16b load_b32x16(const __bf16* p) {
  union { v16b v; uint4 u[2]; } r;
  r.u[0] = *(const uint4*)(p);
  r.u[1] = *(const uint4*)(p + 8);
  return r.v;
}

// ---- Tensor Data Mover: DMA a 2-D tile (bf16) from global into LDS --------
// D# per CDNA5 ISA: group0 {count=1, lds_addr,
// global_addr[56:0], type=2}; group1 {data_size=1(2B), tensor dims, tile dims,
// dim0 stride}; groups 2/3 zero (2-D tensor). Tracked by TENSORcnt.
__device__ __forceinline__ void tdm_load_2d_bf16(
    const void* gtile, unsigned lds_byte_off,
    unsigned tensor_d0, unsigned tensor_d1,
    unsigned tile_d0, unsigned tile_d1, unsigned stride0)
{
  unsigned long long ga = (unsigned long long)gtile;
  u32x4 g0;
  g0.x = 1u;                                             // count=1, user D#
  g0.y = lds_byte_off;                                   // lds_addr (bytes)
  g0.z = (unsigned)(ga & 0xFFFFFFFFull);                 // global_addr[31:0]
  g0.w = (unsigned)((ga >> 32) & 0x1FFFFFFull) | (2u << 30); // [56:32] | type=2
  i32x8 g1;
  g1[0] = (int)(1u << 16);                               // data_size=1 -> 2 bytes
  g1[1] = (int)((tensor_d0 & 0xFFFFu) << 16);            // tensor_dim0[15:0]
  g1[2] = (int)(((tensor_d0 >> 16) & 0xFFFFu) |
                ((tensor_d1 & 0xFFFFu) << 16));          // dim0[31:16] | dim1[15:0]
  g1[3] = (int)(((tensor_d1 >> 16) & 0xFFFFu) |
                ((tile_d0 & 0xFFFFu) << 16));            // dim1[31:16] | tile_dim0
  g1[4] = (int)(tile_d1 & 0xFFFFu);                      // tile_dim1 (tile_dim2=0)
  g1[5] = (int)stride0;                                  // dim0_stride[31:0]
  g1[6] = 0;                                             // stride[47:32]=0, dim1_stride lo=0
  g1[7] = 0;
  i32x4 z4 = {0, 0, 0, 0};
#if defined(__clang_major__) && (__clang_major__ >= 23)
  i32x8 z8 = {0, 0, 0, 0, 0, 0, 0, 0};
  __builtin_amdgcn_tensor_load_to_lds(g0, g1, z4, z4, z8, 0);
#else
  __builtin_amdgcn_tensor_load_to_lds(g0, g1, z4, z4, 0);
#endif
}

// ---- weight conversion ----------------------------------------------------
__global__ __launch_bounds__(256) void k_cvt(const float* __restrict__ s,
                                             __bf16* __restrict__ d, int n) {
  int i = blockIdx.x * blockDim.x + threadIdx.x;
  int stride = gridDim.x * blockDim.x;
  for (; i < n; i += stride) d[i] = (__bf16)s[i];
}

// ---- embedding gather + dilated depthwise conv + LN + SiLU ----------------
__global__ __launch_bounds__(256) void k_embed_conv(
    const int* __restrict__ hashes, const int* __restrict__ offs,
    const float* __restrict__ emb,  const float* __restrict__ convw,
    const float* __restrict__ lncg, const float* __restrict__ lncb,
    __bf16* __restrict__ Ebf)
{
  __shared__ float etile[(TCHUNK + HALO) * EDIM];   // 41*256 f32
  __shared__ float ctile[TCHUNK * EDIM];            // 32*256 f32
  int tid  = threadIdx.x;
  int lane = tid & 31, wid = tid >> 5;
  int b  = blockIdx.x / (SEQ / TCHUNK);
  int t0 = (blockIdx.x % (SEQ / TCHUNK)) * TCHUNK;

  for (int task = tid; task < (TCHUNK + HALO) * NHEADS; task += 256) {
    int r = task >> 3;
    int h = task & 7;
    int tg = t0 - HALO + r;
    float4* dst = (float4*)&etile[r * EDIM + h * HDIM];
    if (tg < 0) {
      float4 z = make_float4(0.f, 0.f, 0.f, 0.f);
      for (int j = 0; j < 8; ++j) dst[j] = z;
    } else {
      int idx = hashes[((size_t)(b * SEQ + tg)) * NHEADS + h] + offs[h];
      const float4* src = (const float4*)(emb + (size_t)idx * HDIM);
      for (int j = 0; j < 8; ++j) dst[j] = src[j];
    }
  }
  __syncthreads();

  {
    int c = tid;
    float w0 = convw[c * 4 + 0], w1 = convw[c * 4 + 1];
    float w2 = convw[c * 4 + 2], w3 = convw[c * 4 + 3];
    for (int tt = 0; tt < TCHUNK; ++tt) {
      ctile[tt * EDIM + c] = etile[(tt + 0) * EDIM + c] * w0
                           + etile[(tt + 3) * EDIM + c] * w1
                           + etile[(tt + 6) * EDIM + c] * w2
                           + etile[(tt + 9) * EDIM + c] * w3;
    }
  }
  __syncthreads();

  for (int tt = wid * 4; tt < wid * 4 + 4; ++tt) {
    float s = 0.f, s2 = 0.f;
    for (int j = 0; j < 8; ++j) {
      float v = ctile[tt * EDIM + lane + 32 * j];
      s += v; s2 += v * v;
    }
    for (int m = 16; m > 0; m >>= 1) { s += __shfl_xor(s, m); s2 += __shfl_xor(s2, m); }
    float mean = s * (1.f / EDIM);
    float var  = s2 * (1.f / EDIM) - mean * mean;
    float rstd = rsqrtf(var + 1e-5f);
    __bf16* dst = Ebf + ((size_t)(b * SEQ + t0 + tt)) * EDIM;
    for (int j = 0; j < 8; ++j) {
      int c = lane + 32 * j;
      float y  = (ctile[tt * EDIM + c] - mean) * rstd * lncg[c] + lncb[c];
      float si = y / (1.f + expf(-y));                      // silu
      dst[c] = (__bf16)(etile[(tt + HALO) * EDIM + c] + si);
    }
  }
}

// ---- key GEMM -> gate -> value GEMM (fused, per 16-row tile) --------------
__global__ __launch_bounds__(256) void k_kv_gate(
    const __bf16* __restrict__ Ebf, const __bf16* __restrict__ WkB,
    const __bf16* __restrict__ WvB, const float* __restrict__ x,
    const float* __restrict__ lnkg, const float* __restrict__ lnkb,
    const float* __restrict__ lnqg, const float* __restrict__ lnqb,
    __bf16* __restrict__ Vg)
{
  __shared__ __bf16 asmem[16 * EDIM];      // 8 KB  A tile
  __shared__ float  keybuf[16 * HID];      // 128 KB key tile
  __shared__ float  gbuf[16];
  int tid = threadIdx.x, lane = tid & 31, wid = tid >> 5;
  int hi = lane >> 4, nl = lane & 15;
  size_t m0 = (size_t)blockIdx.x * 16;

  {
    const unsigned int* src = (const unsigned int*)(Ebf + m0 * EDIM);
    unsigned int* dst = (unsigned int*)asmem;
    for (int i = tid; i < 16 * EDIM / 2; i += 256) dst[i] = src[i];
  }
  __syncthreads();

  // key = E @ Wk^T
  for (int nt = wid; nt < HID / 16; nt += 8) {
    int n0 = nt * 16;
    v8f acc = {};
    const __bf16* brow = WkB + (size_t)(n0 + nl) * EDIM + (hi << 4);
    for (int kk = 0; kk < EDIM / 32; ++kk) {
      v16b a  = load_a16x32(asmem, EDIM, kk * 32, lane);
      v16b bb = load_b32x16(brow + kk * 32);
      acc = __builtin_amdgcn_wmma_f32_16x16x32_bf16(false, a, false, bb,
                                                    (short)0, acc, false, false);
    }
    for (int r = 0; r < 8; ++r) keybuf[(r + 8 * hi) * HID + n0 + nl] = acc[r];
  }
  __syncthreads();

  // per-row gate: g = sigmoid(signed_sqrt( LN(key) . LN(x) / sqrt(HID) ))
  for (int m = wid * 2; m < wid * 2 + 2; ++m) {
    size_t rg = m0 + m;
    float s = 0.f, s2 = 0.f;
    for (int j = 0; j < HID / 32; ++j) {
      float v = keybuf[m * HID + lane + 32 * j]; s += v; s2 += v * v;
    }
    for (int k = 16; k > 0; k >>= 1) { s += __shfl_xor(s, k); s2 += __shfl_xor(s2, k); }
    float km = s / (float)HID, kvv = s2 / (float)HID - km * km;
    float kr = rsqrtf(kvv + 1e-5f);

    const float* xr = x + rg * HID;
    s = 0.f; s2 = 0.f;
    for (int j = 0; j < HID / 32; ++j) {
      float v = xr[lane + 32 * j]; s += v; s2 += v * v;
    }
    for (int k = 16; k > 0; k >>= 1) { s += __shfl_xor(s, k); s2 += __shfl_xor(s2, k); }
    float qm = s / (float)HID, qvv = s2 / (float)HID - qm * qm;
    float qr = rsqrtf(qvv + 1e-5f);

    float d = 0.f;
    for (int j = 0; j < HID / 32; ++j) {
      int c = lane + 32 * j;
      float kn = (keybuf[m * HID + c] - km) * kr * lnkg[c] + lnkb[c];
      float qn = (xr[c] - qm) * qr * lnqg[c] + lnqb[c];
      d += kn * qn;
    }
    for (int k = 16; k > 0; k >>= 1) d += __shfl_xor(d, k);
    if (lane == 0) {
      float gg = d * 0.022097086912079608f;   // 1/sqrt(2048)
      float a  = sqrtf(fmaxf(fabsf(gg), 1e-6f));
      float sg = copysignf(a, gg);
      gbuf[m] = 1.f / (1.f + expf(-sg));
    }
  }
  __syncthreads();

  // value = g * (E @ Wv^T), stored bf16 for the final GEMM
  for (int nt = wid; nt < HID / 16; nt += 8) {
    int n0 = nt * 16;
    v8f acc = {};
    const __bf16* brow = WvB + (size_t)(n0 + nl) * EDIM + (hi << 4);
    for (int kk = 0; kk < EDIM / 32; ++kk) {
      v16b a  = load_a16x32(asmem, EDIM, kk * 32, lane);
      v16b bb = load_b32x16(brow + kk * 32);
      acc = __builtin_amdgcn_wmma_f32_16x16x32_bf16(false, a, false, bb,
                                                    (short)0, acc, false, false);
    }
    for (int r = 0; r < 8; ++r) {
      int m = r + 8 * hi;
      Vg[(m0 + m) * HID + n0 + nl] = (__bf16)(acc[r] * gbuf[m]);
    }
  }
}

// ---- out = Vg @ Wo^T  (M=16384, N=2048, K=2048) ---------------------------
// 128x128 block tiles; K stepped by 32 with TDM double-buffered LDS staging.
// LDS layout (single shared array -> LDS offset 0):
//   A tiles: [buf][128 rows][32 k] bf16 at byte 0 / 8192
//   B tiles: [buf][128 nrow][32 k] bf16 at byte 16384 / 24576
#define A_OFF(buf) ((unsigned)(buf) * 8192u)
#define B_OFF(buf) (16384u + (unsigned)(buf) * 8192u)

__global__ __launch_bounds__(256) void k_out_gemm(
    const __bf16* __restrict__ Vg, const __bf16* __restrict__ WoB,
    float* __restrict__ out)
{
  __shared__ __bf16 smem[16384];   // 32 KB: 2x(A 8KB) + 2x(B 8KB)
  int tid = threadIdx.x, lane = tid & 31, wid = tid >> 5;
  int hi = lane >> 4, nl = lane & 15;
  int wm = wid & 1;                 // 0..1 : 64-row half
  int wn = wid >> 1;                // 0..3 : 32-col slice
  size_t m_blk = (size_t)blockIdx.x * 128;
  size_t n_blk = (size_t)blockIdx.y * 128;

  const __bf16* Atile = Vg  + m_blk * HID;   // 128 x 2048, row-major
  const __bf16* Btile = WoB + n_blk * HID;   // 128 x 2048, row-major

  v8f acc[4][2];
  for (int i = 0; i < 4; ++i)
    for (int j = 0; j < 2; ++j)
      acc[i][j] = (v8f){};

  const int KSTEPS = HID / 32;     // 64

  // prologue: DMA k-chunk 0 into buffer 0 (wave 0 drives the TDM)
  if (wid == 0) {
    tdm_load_2d_bf16(Atile, A_OFF(0), HID, (unsigned)BT, 32, 128, HID);
    tdm_load_2d_bf16(Btile, B_OFF(0), HID, (unsigned)HID, 32, 128, HID);
  }

  for (int ks = 0; ks < KSTEPS; ++ks) {
    int buf = ks & 1;
    __syncthreads();               // everyone done reading buf^1 from prev iter
    if (wid == 0) {
      if (ks + 1 < KSTEPS) {
        int k0 = (ks + 1) * 32;
        tdm_load_2d_bf16(Atile + k0, A_OFF(buf ^ 1), HID, (unsigned)BT, 32, 128, HID);
        tdm_load_2d_bf16(Btile + k0, B_OFF(buf ^ 1), HID, (unsigned)HID, 32, 128, HID);
        __builtin_amdgcn_s_wait_tensorcnt(2);   // current buf's 2 DMAs done
      } else {
        __builtin_amdgcn_s_wait_tensorcnt(0);
      }
    }
    __syncthreads();               // buf is ready for all waves

    const __bf16* abase = smem + (A_OFF(buf) >> 1);
    const __bf16* bbase = smem + (B_OFF(buf) >> 1);
    v16b bfr[2];
    for (int j = 0; j < 2; ++j)
      bfr[j] = load_b32x16(bbase + (size_t)(wn * 32 + j * 16 + nl) * 32 + (hi << 4));
    for (int i = 0; i < 4; ++i) {
      v16b a = load_a16x32(abase + (size_t)(wm * 64 + i * 16) * 32, 32, 0, lane);
      for (int j = 0; j < 2; ++j)
        acc[i][j] = __builtin_amdgcn_wmma_f32_16x16x32_bf16(
            false, a, false, bfr[j], (short)0, acc[i][j], false, false);
    }
  }

  for (int i = 0; i < 4; ++i)
    for (int j = 0; j < 2; ++j)
      for (int r = 0; r < 8; ++r) {
        size_t row = m_blk + wm * 64 + i * 16 + r + 8 * hi;
        out[row * HID + n_blk + wn * 32 + j * 16 + nl] = acc[i][j][r];
      }
}

// ---------------------------------------------------------------------------
extern "C" void kernel_launch(void* const* d_in, const int* in_sizes, int n_in,
                              void* d_out, int out_size, void* d_ws, size_t ws_size,
                              hipStream_t stream) {
  const float* x      = (const float*)d_in[0];
  const int*   hashes = (const int*)d_in[1];
  const int*   offs   = (const int*)d_in[2];
  const float* emb    = (const float*)d_in[3];
  const float* convw  = (const float*)d_in[4];
  const float* lncg   = (const float*)d_in[5];
  const float* lncb   = (const float*)d_in[6];
  const float* Wk     = (const float*)d_in[7];
  const float* Wv     = (const float*)d_in[8];
  const float* Wo     = (const float*)d_in[9];
  const float* lnkg   = (const float*)d_in[10];
  const float* lnkb   = (const float*)d_in[11];
  const float* lnqg   = (const float*)d_in[12];
  const float* lnqb   = (const float*)d_in[13];
  float* out = (float*)d_out;

  // workspace layout (bf16 activations/weights), ~86 MB total
  char* ws = (char*)d_ws;
  __bf16* Ebf = (__bf16*)ws;  ws += (size_t)BT * EDIM * 2;    // 8 MB
  __bf16* WkB = (__bf16*)ws;  ws += (size_t)HID * EDIM * 2;   // 1 MB
  __bf16* WvB = (__bf16*)ws;  ws += (size_t)HID * EDIM * 2;   // 1 MB
  __bf16* WoB = (__bf16*)ws;  ws += (size_t)HID * HID * 2;    // 8 MB
  __bf16* Vg  = (__bf16*)ws;                                  // 64 MB

  k_cvt<<<1024, 256, 0, stream>>>(Wk, WkB, HID * EDIM);
  k_cvt<<<1024, 256, 0, stream>>>(Wv, WvB, HID * EDIM);
  k_cvt<<<4096, 256, 0, stream>>>(Wo, WoB, HID * HID);

  k_embed_conv<<<BATCH * (SEQ / TCHUNK), 256, 0, stream>>>(
      hashes, offs, emb, convw, lncg, lncb, Ebf);

  k_kv_gate<<<BT / 16, 256, 0, stream>>>(
      Ebf, WkB, WvB, x, lnkg, lnkb, lnqg, lnqb, Vg);

  dim3 g3(BT / 128, HID / 128);
  k_out_gemm<<<g3, 256, 0, stream>>>(Vg, WoB, out);
}
